// EdgeModel_19078244729180
// MI455X (gfx1250) — matrix operand order repaired and checked by reading, not compiled, not run on hardware
//
#include <hip/hip_runtime.h>
#include <hip/hip_bf16.h>
#include <stdint.h>

// ---------------------------------------------------------------------------
// Edge-model MLP + LayerNorm for MI455X (gfx1250), bf16 WMMA path.
//   attrs[E,256] = concat(e, x[col], x[row], u[batch[row]])
//   h1  = relu(attrs @ W1 + b1)      (256 -> 128)
//   h2  = relu(h1    @ W2 + b2)      (128 -> 128)
//   out = layernorm(h2) * ln_g + ln_b
// 98.3 GFLOP; ~0.8 GB net DRAM traffic (x gathers hit the 192MB L2).
// GEMMs computed transposed: A = weight^T (LDS, per-lane K split in two 8-wide chunks),
// B = activations^T (lane = edge column). Wave w owns edges [16w,16w+16).
// Weights are pre-converted to a bf16 LDS image in d_ws by a prep kernel,
// then DMA'd into LDS per block with TENSOR_LOAD_TO_LDS (TDM, TENSORcnt).
// ---------------------------------------------------------------------------

#define ATTR_S 272   // u16 per attrs row (256 + 16 pad)  = 544 B (16B mult)
#define W1_S   272
#define H1_S   136   // u16 per h1 row   (128 + 8 pad)    = 272 B (16B mult)
#define W2_S   136
#define WIMG_U16  (128 * W1_S + 128 * W2_S)   // 52224 u16 = 104448 B
#define WIMG_DW   ((WIMG_U16 * 2) / 4)        // 26112 dwords

typedef unsigned short u16;
typedef u16   u16x4  __attribute__((ext_vector_type(4)));
typedef u16   u16x8  __attribute__((ext_vector_type(8)));
typedef __bf16 bf16_t;
typedef bf16_t v16bf __attribute__((ext_vector_type(16)));
typedef float  v8f   __attribute__((ext_vector_type(8)));
typedef unsigned int u32x4 __attribute__((ext_vector_type(4)));
typedef int   i32x4  __attribute__((ext_vector_type(4)));
typedef int   i32x8  __attribute__((ext_vector_type(8)));

#if __has_builtin(__builtin_amdgcn_tensor_load_to_lds) && \
    __has_builtin(__builtin_amdgcn_s_wait_tensorcnt)
#define HAVE_TDM 1
#else
#define HAVE_TDM 0
#endif

union FragU { u16x8 h[2]; v16bf v; };

__device__ __forceinline__ u16 f2bf(float f) {
    unsigned u = __float_as_uint(f);
    u += 0x7FFFu + ((u >> 16) & 1u);      // round-to-nearest-even
    return (u16)(u >> 16);
}

__device__ __forceinline__ v16bf ld_contig(const u16* p) {     // 16 contiguous bf16
    FragU f;
    f.h[0] = *(const u16x8*)(p);
    f.h[1] = *(const u16x8*)(p + 8);
    return f.v;
}
__device__ __forceinline__ v16bf ld_split(const u16* p0, const u16* p1) { // A layout
    FragU f;
    f.h[0] = *(const u16x8*)(p0);
    f.h[1] = *(const u16x8*)(p1);
    return f.v;
}

__device__ __forceinline__ void stage_row64(u16* dst, const float* __restrict__ src) {
#pragma unroll
    for (int j = 0; j < 64; j += 4) {
        float4 v = *(const float4*)(src + j);
        u16x4 p = { f2bf(v.x), f2bf(v.y), f2bf(v.z), f2bf(v.w) };
        *(u16x4*)(dst + j) = p;
    }
}

// ---------------------------------------------------------------------------
// Prep: convert W1/W2 fp32 -> bf16 image in d_ws, laid out exactly as the
// LDS tiles the main kernel wants: W1^T[128][W1_S] then W2^T[128][W2_S].
// ---------------------------------------------------------------------------
__global__ __launch_bounds__(256)
void weight_prep_kernel(const float* __restrict__ W1, const float* __restrict__ W2,
                        u16* __restrict__ wimg) {
    const int i = blockIdx.x * 256 + threadIdx.x;
    if (i < 256 * 128) {                       // W1[k][n], k<256, n<128
        const int k = i >> 7, n = i & 127;
        wimg[n * W1_S + k] = f2bf(W1[i]);
    } else if (i < 256 * 128 + 128 * 128) {    // W2[k][n]
        const int j = i - 256 * 128;
        const int k = j >> 7, n = j & 127;
        wimg[128 * W1_S + n * W2_S + k] = f2bf(W2[j]);
    }
}

__global__ __launch_bounds__(256, 1)
void edge_mlp_ln_kernel(const float* __restrict__ x,  const float* __restrict__ e,
                        const float* __restrict__ u,
                        const float* __restrict__ W1, const float* __restrict__ b1,
                        const float* __restrict__ W2, const float* __restrict__ b2,
                        const float* __restrict__ lng, const float* __restrict__ lnb,
                        const int*   __restrict__ eidx,   // [2,E] flat: row then col
                        const int*   __restrict__ batch,  // [N]
                        const u16*   __restrict__ wbf,    // bf16 weight image or null
                        float* __restrict__ out, int E)
{
    extern __shared__ char smemRaw[];
    float* sB1  = (float*)smemRaw;               // 128
    float* sB2  = sB1 + 128;
    float* sG   = sB2 + 128;
    float* sLB  = sG  + 128;                     // 2048 B of f32 params
    u16*   sW1T = (u16*)(sLB + 128);             // [128][W1_S]  W1^T bf16 @ LDS+2048
    u16*   sW2T = sW1T + 128 * W1_S;             // [128][W2_S]  W2^T
    u16*   sAttr= sW2T + 128 * W2_S;             // [128 edges][ATTR_S]
    u16*   sH1  = sAttr + 128 * ATTR_S;          // [128 edges][H1_S]

    const int t = threadIdx.x;
    const int blockBase = blockIdx.x * 128;

    // ---- stage weights ----
    if (wbf) {
#if HAVE_TDM
        if (t < 32) {   // TDM op is per-wave: only wave 0 issues the DMA
            const unsigned long long ga = (unsigned long long)(uintptr_t)wbf;
            u32x4 g0;
            g0[0] = 1u;                                   // count=1, no gather
            g0[1] = 2048u;                                // lds_addr (bytes): sW1T
            g0[2] = (unsigned)(ga & 0xFFFFFFFFu);         // global_addr[31:0]
            g0[3] = (unsigned)((ga >> 32) & 0x1FFFFFFu)   // global_addr[56:32]
                  | (2u << 30);                           // type=2 ("image")
            i32x8 g1;
            g1[0] = (int)(2u << 16);            // data_size=2 -> 4-byte elems
            g1[1] = (int)((WIMG_DW & 0xFFFFu) << 16);   // tensor_dim0[15:0]
            g1[2] = (int)(((WIMG_DW >> 16) & 0xFFFFu)   // tensor_dim0[31:16]
                  | (1u << 16));                        // tensor_dim1 = 1
            g1[3] = (int)((WIMG_DW & 0xFFFFu) << 16);   // tile_dim0 = 26112 dwords
            g1[4] = 1;                                  // tile_dim1=1, tile_dim2=0
            g1[5] = (int)WIMG_DW;                       // tensor_dim0_stride[31:0]
            g1[6] = 0;                                  // stride0 hi, stride1 lo
            g1[7] = 0;
            const i32x4 gz4 = {0, 0, 0, 0};             // groups 2/3 unused (<=2D)
            const i32x8 gz8 = {0, 0, 0, 0, 0, 0, 0, 0}; // 6-arg form extra group
            __builtin_amdgcn_tensor_load_to_lds(g0, g1, gz4, gz4, gz8, 0);
            __builtin_amdgcn_s_wait_tensorcnt(0);
        }
#else
        // prepared bf16 image, plain vectorized LDS fill (W1^T and W2^T contiguous)
        for (int i = t; i < WIMG_U16 / 8; i += 256)
            ((u16x8*)sW1T)[i] = ((const u16x8*)wbf)[i];
#endif
    } else {
        // fallback: convert fp32 weights in-kernel
        for (int i = t; i < 256 * 128; i += 256) {   // W1[k][n]
            int k = i >> 7, n = i & 127;
            sW1T[n * W1_S + k] = f2bf(W1[i]);
        }
        for (int i = t; i < 128 * 128; i += 256) {   // W2[k][n]
            int k = i >> 7, n = i & 127;
            sW2T[n * W2_S + k] = f2bf(W2[i]);
        }
    }
    if (t < 128) { sB1[t] = b1[t]; sB2[t] = b2[t]; sG[t] = lng[t]; sLB[t] = lnb[t]; }

    // ---- stage attrs tile: 2 threads per edge, 128 floats each ----
    {
        int m = t >> 1, half = t & 1;
        int ei = blockBase + m;
        u16* dst = sAttr + m * ATTR_S + half * 128;
        if (ei < E) {
            if (half == 0) {
                int c = eidx[E + ei];                       // receiver index
                stage_row64(dst,      e + (size_t)ei * 64);
                stage_row64(dst + 64, x + (size_t)c  * 64);
            } else {
                int r = eidx[ei];                           // sender index
                stage_row64(dst,      x + (size_t)r * 64);
                stage_row64(dst + 64, u + (size_t)batch[r] * 64);
            }
        } else {
            u16x4 z = {0, 0, 0, 0};
#pragma unroll
            for (int j = 0; j < 128; j += 4) *(u16x4*)(dst + j) = z;
        }
    }
    __syncthreads();

    // ---- per-wave compute ----
    const int wave = t >> 5;
    const int lane = t & 31;
    const int l16  = lane & 15;
    const int hi   = lane >> 4;
    const int hi8  = hi * 8;
    const int hi16 = hi * 16;
    const int m    = wave * 16 + l16;            // this lane's edge (B column)

    // GEMM1: h1^T = W1^T(128x256) @ attrs^T(256 x 16-edge-block)
    const u16* aRow = sAttr + (size_t)m * ATTR_S;
    v16bf bA[8];
#pragma unroll
    for (int kb = 0; kb < 8; ++kb)
        bA[kb] = ld_split(aRow + kb * 32 + hi8, aRow + kb * 32 + 16 + hi8);

#pragma unroll
    for (int nt = 0; nt < 8; ++nt) {
        const int nB = nt * 16;
        v8f acc;
#pragma unroll
        for (int r = 0; r < 8; ++r) acc[r] = sB1[nB + hi8 + r];
        const u16* w = sW1T + (size_t)(nB + l16) * W1_S;
#pragma unroll
        for (int kb = 0; kb < 8; ++kb) {
            v16bf a = ld_split(w + kb * 32 + hi8, w + kb * 32 + 16 + hi8);
            acc = __builtin_amdgcn_wmma_f32_16x16x32_bf16(
                false, a, false, bA[kb], (short)0, acc, false, false);
        }
        u16x8 o;                                  // relu -> bf16
#pragma unroll
        for (int r = 0; r < 8; ++r) {
            float v = acc[r];
            o[r] = f2bf(v > 0.f ? v : 0.f);
        }
        *(u16x8*)(sH1 + (size_t)m * H1_S + nB + hi8) = o;   // one b128 store
    }

    // GEMM2: out^T = W2^T(128x128) @ h1^T(128 x 16-edge-block)
    const u16* hRow = sH1 + (size_t)m * H1_S;
    v16bf bH[4];
#pragma unroll
    for (int kb = 0; kb < 4; ++kb)
        bH[kb] = ld_split(hRow + kb * 32 + hi8, hRow + kb * 32 + 16 + hi8);

    v8f acc2[8];
#pragma unroll
    for (int nt = 0; nt < 8; ++nt) {
        const int nB = nt * 16;
#pragma unroll
        for (int r = 0; r < 8; ++r) acc2[nt][r] = sB2[nB + hi8 + r];
        const u16* w = sW2T + (size_t)(nB + l16) * W2_S;
#pragma unroll
        for (int kb = 0; kb < 4; ++kb) {
            v16bf a = ld_split(w + kb * 32 + hi8, w + kb * 32 + 16 + hi8);
            acc2[nt] = __builtin_amdgcn_wmma_f32_16x16x32_bf16(
                false, a, false, bH[kb], (short)0, acc2[nt], false, false);
        }
    }

    // ---- relu + layernorm over the 128 outputs of edge m ----
    float s = 0.f, sq = 0.f;
#pragma unroll
    for (int nt = 0; nt < 8; ++nt)
#pragma unroll
        for (int r = 0; r < 8; ++r) {
            float v = acc2[nt][r];
            v = v > 0.f ? v : 0.f;               // cmp+cndmask, dual-issuable
            acc2[nt][r] = v;
            s  += v;
            sq += v * v;
        }
    // lanes L and L+16 hold the two halves of the same edge
    s  += __shfl_xor(s,  16, 32);
    sq += __shfl_xor(sq, 16, 32);
    const float mu  = s * (1.f / 128.f);
    const float var = sq * (1.f / 128.f) - mu * mu;
    const float rs  = rsqrtf(var + 1e-5f);

    const int ei = blockBase + m;
    if (ei < E) {
        float* op = out + (size_t)ei * 128;
#pragma unroll
        for (int nt = 0; nt < 8; ++nt) {
            const int nB = nt * 16 + hi8;
            float4 lo, hiv;
            lo.x  = (acc2[nt][0] - mu) * rs * sG[nB + 0] + sLB[nB + 0];
            lo.y  = (acc2[nt][1] - mu) * rs * sG[nB + 1] + sLB[nB + 1];
            lo.z  = (acc2[nt][2] - mu) * rs * sG[nB + 2] + sLB[nB + 2];
            lo.w  = (acc2[nt][3] - mu) * rs * sG[nB + 3] + sLB[nB + 3];
            hiv.x = (acc2[nt][4] - mu) * rs * sG[nB + 4] + sLB[nB + 4];
            hiv.y = (acc2[nt][5] - mu) * rs * sG[nB + 5] + sLB[nB + 5];
            hiv.z = (acc2[nt][6] - mu) * rs * sG[nB + 6] + sLB[nB + 6];
            hiv.w = (acc2[nt][7] - mu) * rs * sG[nB + 7] + sLB[nB + 7];
            *(float4*)(op + nB)     = lo;
            *(float4*)(op + nB + 4) = hiv;
        }
    }
}

extern "C" void kernel_launch(void* const* d_in, const int* in_sizes, int n_in,
                              void* d_out, int out_size, void* d_ws, size_t ws_size,
                              hipStream_t stream) {
    const float* x   = (const float*)d_in[0];
    const float* e   = (const float*)d_in[1];
    const float* u   = (const float*)d_in[2];
    const float* W1  = (const float*)d_in[3];
    const float* b1  = (const float*)d_in[4];
    const float* W2  = (const float*)d_in[5];
    const float* b2  = (const float*)d_in[6];
    const float* lng = (const float*)d_in[7];
    const float* lnb = (const float*)d_in[8];
    const int*   ei  = (const int*)d_in[9];
    const int*   bat = (const int*)d_in[10];
    float* out = (float*)d_out;

    const int E = in_sizes[1] / 64;              // e is [E,64]

    const u16* wbf = nullptr;
    if (ws_size >= (size_t)WIMG_U16 * sizeof(u16)) {
        const int total = 256 * 128 + 128 * 128;          // 49152 elements
        weight_prep_kernel<<<(total + 255) / 256, 256, 0, stream>>>(W1, W2, (u16*)d_ws);
        wbf = (const u16*)d_ws;
    }

    const size_t smem = 4 * 128 * sizeof(float) +
        (size_t)(128 * W1_S + 128 * W2_S + 128 * ATTR_S + 128 * H1_S) * sizeof(u16);
    // ~211 KB dynamic LDS (one workgroup per WGP; 320 KB available)
    (void)hipFuncSetAttribute((const void*)edge_mlp_ln_kernel,
                              hipFuncAttributeMaxDynamicSharedMemorySize, (int)smem);
    const int grid = (E + 127) / 128;
    edge_mlp_ln_kernel<<<grid, 256, smem, stream>>>(
        x, e, u, W1, b1, W2, b2, lng, lnb, ei, bat, wbf, out, E);
}
